// GRUDecoder_56573309224617
// MI455X (gfx1250) — matrix operand order repaired and checked by reading, not compiled, not run on hardware
//
#include <hip/hip_runtime.h>
#include <hip/hip_bf16.h>
#include <math.h>

// ---------------------------------------------------------------------------
// GRU decoder for MI455X (gfx1250, wave32, WMMA).
// fp16 weights converted once (91 MB -> resident in the 192 MB L2); all
// matmuls use v_wmma_f32_16x16x32_f16. 32 sequential decode steps.
//   - k_gru: 8 waves/block, K split 4 ways, LDS f32 accumulation (ds_add_f32)
//   - k_outgemm: 4 N-tiles per wave to amortize the A (H1) loads 4x
// ---------------------------------------------------------------------------

#define EMB   1024
#define HID   1024
#define VOCAB 32000
#define BATCH 32
#define TSTEPS 32

typedef __attribute__((ext_vector_type(16))) _Float16 v16h;
typedef __attribute__((ext_vector_type(8)))  float    v8f;

union HVec {
    uint4 u[2];   // 2 x 16 bytes = 32 bytes
    v16h  v;      // 16 halves
};

__device__ __forceinline__ v8f wmma_f16(const v16h& a, const v16h& b, const v8f& c) {
    return __builtin_amdgcn_wmma_f32_16x16x32_f16(
        /*neg_a=*/false, a, /*neg_b=*/false, b,
        /*c_mod=*/(short)0, c, /*reuse_a=*/false, /*reuse_b=*/false);
}

// --------------------------- one-time kernels ------------------------------

__global__ void k_f32_to_f16(const float* __restrict__ s, _Float16* __restrict__ d,
                             unsigned long long n) {
    unsigned long long i = (unsigned long long)blockIdx.x * blockDim.x + threadIdx.x;
    if (i < n) d[i] = (_Float16)s[i];
}

__global__ void k_init(const float* __restrict__ enc,
                       float* __restrict__ h0f, float* __restrict__ h1f,
                       _Float16* __restrict__ h0h, _Float16* __restrict__ h1h,
                       int* __restrict__ seq, const int* __restrict__ sos) {
    int i = blockIdx.x * blockDim.x + threadIdx.x;
    if (i < BATCH * HID) {
        float v = enc[i];
        h0f[i] = v; h1f[i] = v;
        h0h[i] = (_Float16)v; h1h[i] = (_Float16)v;
    }
    if (i < BATCH) seq[i] = *sos;
}

// --------------------------- per-step kernels ------------------------------

// Gather embedding row for each batch element, convert to fp16 (GEMM A operand).
__global__ void k_embed(const float* __restrict__ emb, const int* __restrict__ seq,
                        _Float16* __restrict__ e) {
    int b = blockIdx.x;                  // 0..31
    int tok = seq[b];
    const float* src = emb + (unsigned long long)tok * EMB;
    for (int i = threadIdx.x; i < EMB; i += blockDim.x)
        e[b * EMB + i] = (_Float16)src[i];
}

// Fused GRU cell for a 16-column slice of the hidden state.
// Block = 256 threads (8 waves). Wave w: matrix (w&1) [0: gi = X@Wih^T,
// 1: gh = H@Whh^T], K-chunk (w>>2? no: w>>1) of 256. Partial 32x16 tiles for
// gates (r,z,n) are accumulated into LDS via ds_add_f32, then gate math.
// grid.x = HID/16 = 64.
__global__ __launch_bounds__(256) void k_gru(
        const _Float16* __restrict__ X,    // [32,1024] fp16 (layer input)
        const _Float16* __restrict__ Hh,   // [32,1024] fp16 (prev hidden)
        const float*    __restrict__ Hf,   // [32,1024] fp32 (prev hidden)
        const _Float16* __restrict__ Wih,  // [3072,1024] fp16 row-major
        const _Float16* __restrict__ Whh,  // [3072,1024] fp16 row-major
        const float*    __restrict__ bih,  // [3072]
        const float*    __restrict__ bhh,  // [3072]
        float*    __restrict__ HfOut,      // [32,1024] fp32 (new hidden)
        _Float16* __restrict__ HhOut) {    // [32,1024] fp16 (new hidden)
    __shared__ float gbuf[2][3][BATCH][16];   // 12 KB

    const int tid  = threadIdx.x;
    const int wave = tid >> 5;           // 0..7
    const int lane = tid & 31;
    const int mat  = wave & 1;           // 0 -> gi, 1 -> gh
    const int kc   = wave >> 1;          // 0..3 -> K chunk of 256
    const int n16  = lane & 15;
    const int hiHalf = lane >> 4;        // 0: lanes 0-15, 1: lanes 16-31
    const int jt = blockIdx.x;           // column tile 0..63

    // zero the LDS accumulators (3072 floats / 256 threads = 12 each)
    {
        float* gz = &gbuf[0][0][0][0];
        for (int i = tid; i < 2 * 3 * BATCH * 16; i += 256) gz[i] = 0.0f;
    }
    __syncthreads();

    const _Float16* A = mat ? Hh : X;
    const _Float16* W = mat ? Whh : Wih;

    v8f acc[3][2];
    const v8f vzero = {0.f, 0.f, 0.f, 0.f, 0.f, 0.f, 0.f, 0.f};
#pragma unroll
    for (int g = 0; g < 3; ++g) { acc[g][0] = vzero; acc[g][1] = vzero; }

    // ISA lane layouts (16-bit, wave32):
    //   A 16x32: lane holds row M=lane&15; K chunks at kb and kb+16, kb = (lane<16)?0:8
    //   B 32x16: lane holds col N=lane&15; K chunks at kb and kb+16, kb = (lane<16)?0:8
    const int aKb = hiHalf * 8;
    const int bKb = hiHalf * 8;
    const int kBeg = kc * 256;
    const int kEnd = kBeg + 256;

    for (int k0 = kBeg; k0 < kEnd; k0 += 32) {
        HVec a_lo, a_hi;
        const _Float16* ap0 = A + (unsigned long long)n16 * EMB + k0 + aKb;
        a_lo.u[0] = *(const uint4*)(ap0);
        a_lo.u[1] = *(const uint4*)(ap0 + 16);
        const _Float16* ap1 = A + (unsigned long long)(n16 + 16) * EMB + k0 + aKb;
        a_hi.u[0] = *(const uint4*)(ap1);
        a_hi.u[1] = *(const uint4*)(ap1 + 16);
#pragma unroll
        for (int g = 0; g < 3; ++g) {
            const _Float16* wp = W + (unsigned long long)(g * HID + jt * 16 + n16) * EMB + k0 + bKb;
            HVec bv;
            bv.u[0] = *(const uint4*)(wp);
            bv.u[1] = *(const uint4*)(wp + 16);
            acc[g][0] = wmma_f16(a_lo.v, bv.v, acc[g][0]);
            acc[g][1] = wmma_f16(a_hi.v, bv.v, acc[g][1]);
        }
    }

    // C/D layout: VGPR r, lane -> row = r + (lane>=16 ? 8 : 0), col = lane&15.
    // Accumulate partial K-chunk results into LDS (ds_add_f32).
    const int rowBase = hiHalf * 8;
#pragma unroll
    for (int g = 0; g < 3; ++g)
#pragma unroll
        for (int mt = 0; mt < 2; ++mt)
#pragma unroll
            for (int r = 0; r < 8; ++r)
                atomicAdd(&gbuf[mat][g][mt * 16 + rowBase + r][n16], acc[g][mt][r]);
    __syncthreads();

    // Gate math: 32x16 elements, 256 threads -> 2 each.
    for (int i = tid; i < BATCH * 16; i += 256) {
        int b = i >> 4;
        int c = i & 15;
        int j = jt * 16 + c;
        float gi_r = gbuf[0][0][b][c] + bih[j];
        float gh_r = gbuf[1][0][b][c] + bhh[j];
        float gi_z = gbuf[0][1][b][c] + bih[HID + j];
        float gh_z = gbuf[1][1][b][c] + bhh[HID + j];
        float gi_n = gbuf[0][2][b][c] + bih[2 * HID + j];
        float gh_n = gbuf[1][2][b][c] + bhh[2 * HID + j];
        float r = 1.0f / (1.0f + __expf(-(gi_r + gh_r)));
        float z = 1.0f / (1.0f + __expf(-(gi_z + gh_z)));
        float n = tanhf(gi_n + r * gh_n);
        float hv = Hf[b * HID + j];
        float hn = (1.0f - z) * n + z * hv;
        HfOut[b * HID + j] = hn;
        HhOut[b * HID + j] = (_Float16)hn;
    }
}

// Output projection: logits[32, 32000] = H1 @ Wo^T + bo.
// One wave per 64-column chunk (4 tiles of 16) so each A load feeds 8 WMMAs.
// grid.x = VOCAB/64 = 500.
__global__ __launch_bounds__(32) void k_outgemm(
        const _Float16* __restrict__ H1,   // [32,1024] fp16
        const _Float16* __restrict__ Wo,   // [32000,1024] fp16 row-major
        const float*    __restrict__ bo,   // [32000]
        float* __restrict__ out) {         // [32,32000] fp32 (this step's slab)
    const int lane = threadIdx.x & 31;
    const int n16  = lane & 15;
    const int hiHalf = lane >> 4;
    const int n0 = blockIdx.x * 64;

    const v8f vzero = {0.f, 0.f, 0.f, 0.f, 0.f, 0.f, 0.f, 0.f};
    v8f acc[4][2];
#pragma unroll
    for (int nt = 0; nt < 4; ++nt) { acc[nt][0] = vzero; acc[nt][1] = vzero; }

    const int aKb = hiHalf * 8;
    const int bKb = hiHalf * 8;

    for (int k0 = 0; k0 < HID; k0 += 32) {
        HVec a_lo, a_hi;
        const _Float16* ap0 = H1 + (unsigned long long)n16 * HID + k0 + aKb;
        a_lo.u[0] = *(const uint4*)(ap0);
        a_lo.u[1] = *(const uint4*)(ap0 + 16);
        const _Float16* ap1 = H1 + (unsigned long long)(n16 + 16) * HID + k0 + aKb;
        a_hi.u[0] = *(const uint4*)(ap1);
        a_hi.u[1] = *(const uint4*)(ap1 + 16);
#pragma unroll
        for (int nt = 0; nt < 4; ++nt) {
            const _Float16* wp =
                Wo + (unsigned long long)(n0 + nt * 16 + n16) * HID + k0 + bKb;
            HVec bv;
            bv.u[0] = *(const uint4*)(wp);
            bv.u[1] = *(const uint4*)(wp + 16);
            acc[nt][0] = wmma_f16(a_lo.v, bv.v, acc[nt][0]);
            acc[nt][1] = wmma_f16(a_hi.v, bv.v, acc[nt][1]);
        }
    }

    const int rowBase = hiHalf * 8;
#pragma unroll
    for (int nt = 0; nt < 4; ++nt) {
        const int ncol = n0 + nt * 16 + n16;
        const float bias = bo[ncol];
#pragma unroll
        for (int r = 0; r < 8; ++r) {
            int m0 = rowBase + r;
            out[(unsigned long long)m0 * VOCAB + ncol]        = acc[nt][0][r] + bias;
            out[(unsigned long long)(m0 + 16) * VOCAB + ncol] = acc[nt][1][r] + bias;
        }
    }
}

// Gumbel-argmax categorical sampling (deterministic hash RNG).
__device__ __forceinline__ unsigned long long splitmix64(unsigned long long z) {
    z += 0x9E3779B97F4A7C15ull;
    z = (z ^ (z >> 30)) * 0xBF58476D1CE4E5B9ull;
    z = (z ^ (z >> 27)) * 0x94D049BB133111EBull;
    return z ^ (z >> 31);
}

__global__ __launch_bounds__(256) void k_sample(const float* __restrict__ logits,
                                                int* __restrict__ seq, int step) {
    __shared__ float s_val[256];
    __shared__ int   s_idx[256];
    const int b = blockIdx.x;
    const int tid = threadIdx.x;

    float best = -3.4e38f;
    int bidx = 0;
    for (int v = tid; v < VOCAB; v += 256) {
        unsigned long long s =
            ((unsigned long long)(unsigned)(step * 0x85EBCA6Bu + 0x12345u) << 32) ^
            ((unsigned long long)(unsigned)b << 20) ^ (unsigned long long)(unsigned)v;
        s = splitmix64(s);
        float u = (float)(s >> 40) * (1.0f / 16777216.0f);   // [0, 1)
        u = fmaxf(u, 1e-12f);
        float g = -__logf(-__logf(u));                        // gumbel
        float sc = logits[(unsigned long long)b * VOCAB + v] + g;
        if (sc > best || (sc == best && v < bidx)) { best = sc; bidx = v; }
    }
    s_val[tid] = best; s_idx[tid] = bidx;
    __syncthreads();
    for (int sft = 128; sft > 0; sft >>= 1) {
        if (tid < sft) {
            float v2 = s_val[tid + sft]; int i2 = s_idx[tid + sft];
            if (v2 > s_val[tid] || (v2 == s_val[tid] && i2 < s_idx[tid])) {
                s_val[tid] = v2; s_idx[tid] = i2;
            }
        }
        __syncthreads();
    }
    if (tid == 0) seq[b] = s_idx[0];
}

// ---------------------------------------------------------------------------

extern "C" void kernel_launch(void* const* d_in, const int* in_sizes, int n_in,
                              void* d_out, int out_size, void* d_ws, size_t ws_size,
                              hipStream_t stream) {
    (void)in_sizes; (void)n_in; (void)out_size; (void)ws_size;

    const float* enc   = (const float*)d_in[0];
    const float* emb   = (const float*)d_in[1];
    const float* Wih0  = (const float*)d_in[2];
    const float* Whh0  = (const float*)d_in[3];
    const float* bih0  = (const float*)d_in[4];
    const float* bhh0  = (const float*)d_in[5];
    const float* Wih1  = (const float*)d_in[6];
    const float* Whh1  = (const float*)d_in[7];
    const float* bih1  = (const float*)d_in[8];
    const float* bhh1  = (const float*)d_in[9];
    const float* Wo    = (const float*)d_in[10];
    const float* bo    = (const float*)d_in[11];
    const int*   sos   = (const int*)d_in[14];
    float* out = (float*)d_out;

    // --- workspace carve (256B aligned) ---
    size_t off = 0;
    auto carve = [&](size_t bytes) -> char* {
        char* p = (char*)d_ws + off;
        off += (bytes + 255) & ~(size_t)255;
        return p;
    };
    const size_t W3H = (size_t)3 * HID * EMB;          // 3,145,728 elems
    _Float16* wih0h = (_Float16*)carve(W3H * 2);
    _Float16* whh0h = (_Float16*)carve(W3H * 2);
    _Float16* wih1h = (_Float16*)carve(W3H * 2);
    _Float16* whh1h = (_Float16*)carve(W3H * 2);
    _Float16* woh   = (_Float16*)carve((size_t)VOCAB * HID * 2);
    float*    h0f[2] = {(float*)carve(BATCH * HID * 4), (float*)carve(BATCH * HID * 4)};
    float*    h1f[2] = {(float*)carve(BATCH * HID * 4), (float*)carve(BATCH * HID * 4)};
    _Float16* h0h[2] = {(_Float16*)carve(BATCH * HID * 2), (_Float16*)carve(BATCH * HID * 2)};
    _Float16* h1h[2] = {(_Float16*)carve(BATCH * HID * 2), (_Float16*)carve(BATCH * HID * 2)};
    _Float16* eh     = (_Float16*)carve(BATCH * EMB * 2);
    int*      seq    = (int*)carve(BATCH * 4);

    // --- one-time: weights fp32 -> fp16 (L2-resident working set) ---
    {
        dim3 blk(256);
        k_f32_to_f16<<<dim3((unsigned)((W3H + 255) / 256)), blk, 0, stream>>>(Wih0, wih0h, W3H);
        k_f32_to_f16<<<dim3((unsigned)((W3H + 255) / 256)), blk, 0, stream>>>(Whh0, whh0h, W3H);
        k_f32_to_f16<<<dim3((unsigned)((W3H + 255) / 256)), blk, 0, stream>>>(Wih1, wih1h, W3H);
        k_f32_to_f16<<<dim3((unsigned)((W3H + 255) / 256)), blk, 0, stream>>>(Whh1, whh1h, W3H);
        size_t nwo = (size_t)VOCAB * HID;
        k_f32_to_f16<<<dim3((unsigned)((nwo + 255) / 256)), blk, 0, stream>>>(Wo, woh, nwo);
        k_init<<<dim3((BATCH * HID + 255) / 256), blk, 0, stream>>>(
            enc, h0f[0], h1f[0], h0h[0], h1h[0], seq, sos);
    }

    // --- sequential decode: 32 steps ---
    for (int t = 0; t < TSTEPS; ++t) {
        const int p = t & 1;
        float* out_t = out + (size_t)t * BATCH * VOCAB;

        k_embed<<<dim3(BATCH), dim3(256), 0, stream>>>(emb, seq, eh);

        k_gru<<<dim3(HID / 16), dim3(256), 0, stream>>>(
            eh, h0h[p], h0f[p], wih0h, whh0h, bih0, bhh0, h0f[p ^ 1], h0h[p ^ 1]);

        k_gru<<<dim3(HID / 16), dim3(256), 0, stream>>>(
            h0h[p ^ 1], h1h[p], h1f[p], wih1h, whh1h, bih1, bhh1, h1f[p ^ 1], h1h[p ^ 1]);

        k_outgemm<<<dim3(VOCAB / 64), dim3(32), 0, stream>>>(h1h[p ^ 1], woh, bo, out_t);

        k_sample<<<dim3(BATCH), dim3(256), 0, stream>>>(out_t, seq, t);
    }
}
